// ZebraMixAttention_60619168415906
// MI455X (gfx1250) — matrix operand order repaired and checked by reading, not compiled, not run on hardware
//
#include <hip/hip_runtime.h>
#include <hip/hip_bf16.h>
#include <stdint.h>

// Problem constants (from reference)
#define BB    2
#define SS    4096
#define HH    2048
#define NH    32
#define NKV   8
#define HD    64
#define BL    256
#define NBLK  (SS / BL)       // 16
#define MROWS (BB * SS)       // 8192
#define KVW   (NKV * HD)      // 512
#define QW    (NH * HD)       // 2048

typedef __bf16          v16bf __attribute__((ext_vector_type(16)));
typedef float           v8f   __attribute__((ext_vector_type(8)));
typedef unsigned short  u16;
typedef unsigned short  us8   __attribute__((ext_vector_type(8)));

union FragAB { us8 u[2]; v16bf v; };   // 32B: 16 bf16 = one WMMA A/B operand
union Pack16 { us8 u[2]; u16 w[16]; };

__device__ __forceinline__ u16 f2bf(float f) {
  union { float f; uint32_t u; } c; c.f = f;
  uint32_t u = c.u;
  u += 0x7FFFu + ((u >> 16) & 1u);     // round-to-nearest-even
  return (u16)(u >> 16);
}
__device__ __forceinline__ float bf2f(u16 h) {
  union { uint32_t u; float f; } c; c.u = ((uint32_t)h) << 16;
  return c.f;
}

// ---------------------------------------------------------------------------
// GEMM: C[M,N] = A[M,K] * W[N,K]^T   (WMMA bf16, f32 accumulate)
// Workgroup 256 threads = 8 waves -> 128x128 C tile; wave = 32x64 sub-tile.
// LDS tiles use 80-byte row stride: 16 consecutive rows of ds_load_b128 hit
// 16 distinct banks (20*r mod 64 is a permutation).
// ---------------------------------------------------------------------------
template<bool A_F32, bool OUT_F32>
__global__ __launch_bounds__(256)
void gemm_wmma(const void* __restrict__ Aptr, const float* __restrict__ W,
               void* __restrict__ Cptr, int M, int N, int K) {
  __shared__ __align__(16) u16 sA[128 * 40];   // [m][k] bf16, stride 40 u16 = 80B
  __shared__ __align__(16) u16 sB[128 * 40];   // [n][k] bf16 (W row-major == B^T)
  const int tid  = threadIdx.x;
  const int lane = tid & 31;
  const int wave = tid >> 5;
  const int wm   = wave & 3;        // 32-row band inside 128
  const int wn   = wave >> 2;       // 64-col band inside 128
  const int g    = lane >> 4;       // lane group (ISA A/B frag layout)
  const int lr   = lane & 15;
  const int m0   = blockIdx.y * 128;
  const int n0   = blockIdx.x * 128;
  const int sr   = tid >> 1;        // staging row 0..127
  const int sc   = (tid & 1) * 16;  // staging col 0 / 16

  v8f acc[2][4] = {};

  for (int kt = 0; kt < K; kt += 32) {
    __syncthreads();
    // --- stage A tile 128x32 -> bf16 ---
    if constexpr (A_F32) {
      const float* a = (const float*)Aptr + (size_t)(m0 + sr) * K + kt + sc;
      Pack16 p;
      for (int i = 0; i < 16; i += 4) {
        float4 v = *(const float4*)(a + i);
        p.w[i] = f2bf(v.x); p.w[i+1] = f2bf(v.y);
        p.w[i+2] = f2bf(v.z); p.w[i+3] = f2bf(v.w);
      }
      *(us8*)&sA[sr*40 + sc]     = p.u[0];
      *(us8*)&sA[sr*40 + sc + 8] = p.u[1];
      if (kt + 32 < K) __builtin_prefetch(a + 32, 0, 1);   // global_prefetch_b8
    } else {
      const u16* a = (const u16*)Aptr + (size_t)(m0 + sr) * K + kt + sc;
      *(us8*)&sA[sr*40 + sc]     = *(const us8*)(a);
      *(us8*)&sA[sr*40 + sc + 8] = *(const us8*)(a + 8);
      if (kt + 32 < K) __builtin_prefetch(a + 32, 0, 1);
    }
    // --- stage B tile (W rows n0..n0+127, k-contiguous) ---
    {
      const float* w = W + (size_t)(n0 + sr) * K + kt + sc;
      Pack16 p;
      for (int i = 0; i < 16; i += 4) {
        float4 v = *(const float4*)(w + i);
        p.w[i] = f2bf(v.x); p.w[i+1] = f2bf(v.y);
        p.w[i+2] = f2bf(v.z); p.w[i+3] = f2bf(v.w);
      }
      *(us8*)&sB[sr*40 + sc]     = p.u[0];
      *(us8*)&sB[sr*40 + sc + 8] = p.u[1];
      if (kt + 32 < K) __builtin_prefetch(w + 32, 0, 1);
    }
    __syncthreads();

    // A frags: lane lr = row; halfwords h<8 -> K=g*8+h, h>=8 -> K=16+g*8+(h-8)
    FragAB afr[2];
    for (int mt = 0; mt < 2; ++mt) {
      int row = wm*32 + mt*16 + lr;
      afr[mt].u[0] = *(const us8*)&sA[row*40 + g*8];
      afr[mt].u[1] = *(const us8*)&sA[row*40 + 16 + g*8];
    }
    // B frags: lane lr = col n; halfwords h<8 -> K=g*8+h, h>=8 -> K=16+g*8+(h-8)
    for (int nt = 0; nt < 4; ++nt) {
      int row = wn*64 + nt*16 + lr;
      FragAB bfr;
      bfr.u[0] = *(const us8*)&sB[row*40 + g*8];
      bfr.u[1] = *(const us8*)&sB[row*40 + 16 + g*8];
      for (int mt = 0; mt < 2; ++mt)
        acc[mt][nt] = __builtin_amdgcn_wmma_f32_16x16x32_bf16(
            false, afr[mt].v, false, bfr.v, (short)0, acc[mt][nt], false, false);
    }
  }

  // C/D layout: VGPR r, lane -> row = r + 8*g, col = lr
  for (int mt = 0; mt < 2; ++mt)
    for (int nt = 0; nt < 4; ++nt) {
      int col = n0 + wn*64 + nt*16 + lr;
      for (int r = 0; r < 8; ++r) {
        int row = m0 + wm*32 + mt*16 + r + 8*g;
        float v = acc[mt][nt][r];
        if constexpr (OUT_F32)
          ((float*)Cptr)[(size_t)row * N + col] = v;
        else
          ((u16*)Cptr)[(size_t)row * N + col] = f2bf(v);
      }
    }
}

// ---------------------------------------------------------------------------
// RoPE, in place on bf16 [rows][nheads*64]; dims d and d+32 share freq.
// ---------------------------------------------------------------------------
__global__ __launch_bounds__(256)
void rope_kernel(u16* __restrict__ buf, int nheads, int total) {
  int idx = blockIdx.x * 256 + threadIdx.x;
  if (idx >= total) return;
  int d   = idx & 31;
  int h   = (idx >> 5) % nheads;
  int row = idx / (32 * nheads);
  int s   = row & (SS - 1);
  float inv = __powf(10000.0f, -(float)d * (1.0f / 32.0f));
  float f = (float)s * inv;
  float sn, cs;
  __sincosf(f, &sn, &cs);
  size_t base = ((size_t)row * nheads + h) * HD + d;
  float x1 = bf2f(buf[base]), x2 = bf2f(buf[base + 32]);
  buf[base]      = f2bf(x1 * cs - x2 * sn);
  buf[base + 32] = f2bf(x2 * cs + x1 * sn);
}

// ---------------------------------------------------------------------------
// Zebra-block flash attention. One workgroup per (head, q-block, batch).
// 8 waves x 32 query rows = 256. Keys streamed in 8 chunks of 64 over the
// [prev block | current block] window. Q/K frags load straight from global
// (b128, frag-segment aligned); V transposed into LDS; P via per-wave LDS.
// ---------------------------------------------------------------------------
__global__ __launch_bounds__(256)
void attn_kernel(const u16* __restrict__ Q, const u16* __restrict__ Kb,
                 const u16* __restrict__ Vb, u16* __restrict__ O) {
  __shared__ __align__(16) u16 sVt[HD * 72];    // [dim][key], stride 72 u16 = 144B
  __shared__ __align__(16) u16 sP[256 * 72];    // [qrow][key], per-wave 32-row bands
  const int h    = blockIdx.x;
  const int blk  = blockIdx.y;
  const int b    = blockIdx.z;
  const int tid  = threadIdx.x, lane = tid & 31, wave = tid >> 5;
  const int g    = lane >> 4, lr = lane & 15;
  const int kvh  = h >> 2;                      // GQA: NH/NKV = 4
  const int q0   = blk * BL;
  const int prevb = (blk == 0) ? 0 : (blk - 1) * BL;

  // Resident Q A-frags: [mt][kstep], rows are dim-contiguous in global
  FragAB qfr[2][2];
  for (int mt = 0; mt < 2; ++mt) {
    int tok = q0 + wave*32 + mt*16 + lr;
    const u16* qrow = Q + ((size_t)(b*SS + tok) * QW + h*HD);
    for (int kk = 0; kk < 2; ++kk) {
      qfr[mt][kk].u[0] = *(const us8*)(qrow + kk*32 + g*8);
      qfr[mt][kk].u[1] = *(const us8*)(qrow + kk*32 + 16 + g*8);
    }
  }

  v8f oacc[2][4] = {};
  float m_run[2][8], l_run[2][8];
  for (int mt = 0; mt < 2; ++mt)
    for (int r = 0; r < 8; ++r) { m_run[mt][r] = -1e30f; l_run[mt][r] = 0.0f; }

  for (int c = 0; c < 8; ++c) {
    __syncthreads();
    // ---- stage V^T chunk: 64 keys x 64 dims, transposed ----
    {
      int key = tid >> 2;
      int seg = (tid & 3) * 16;
      int j   = c*64 + key;
      int tok = (j < BL) ? (prevb + j) : (q0 + j - BL);
      const u16* vrow = Vb + ((size_t)(b*SS + tok) * KVW + kvh*HD + seg);
      Pack16 p;
      p.u[0] = *(const us8*)(vrow);
      p.u[1] = *(const us8*)(vrow + 8);
      for (int i = 0; i < 16; ++i)
        sVt[(seg + i)*72 + key] = p.w[i];
    }
    __syncthreads();

    // ---- scores S = Q K^T (K B-frags direct from global) ----
    v8f sacc[2][4] = {};
    for (int nt = 0; nt < 4; ++nt) {
      int j   = c*64 + nt*16 + lr;
      int tok = (j < BL) ? (prevb + j) : (q0 + j - BL);
      const u16* krow = Kb + ((size_t)(b*SS + tok) * KVW + kvh*HD);
      for (int kk = 0; kk < 2; ++kk) {
        FragAB kf;
        kf.u[0] = *(const us8*)(krow + kk*32 + g*8);
        kf.u[1] = *(const us8*)(krow + kk*32 + 16 + g*8);
        for (int mt = 0; mt < 2; ++mt)
          sacc[mt][nt] = __builtin_amdgcn_wmma_f32_16x16x32_bf16(
              false, qfr[mt][kk].v, false, kf.v, (short)0, sacc[mt][nt], false, false);
      }
    }

    // ---- online softmax (scale, zebra mask, running max/sum) ----
    const float scale = 0.125f;   // 1/sqrt(64)
    for (int mt = 0; mt < 2; ++mt) {
      float mx[8];
      for (int r = 0; r < 8; ++r) {
        int i = wave*32 + mt*16 + r + 8*g;
        float best = -1e30f;
        for (int nt = 0; nt < 4; ++nt) {
          int j = c*64 + nt*16 + lr;
          float s = sacc[mt][nt][r] * scale;
          bool invalid = (j < i) | (j > i + BL) | ((blk == 0) & (j < BL));
          s = invalid ? -1e30f : s;
          sacc[mt][nt][r] = s;
          best = fmaxf(best, s);
        }
        mx[r] = best;
      }
      for (int off = 8; off >= 1; off >>= 1)
        for (int r = 0; r < 8; ++r)
          mx[r] = fmaxf(mx[r], __shfl_xor(mx[r], off, 32));
      float alpha[8], rs[8];
      for (int r = 0; r < 8; ++r) {
        float mn = fmaxf(m_run[mt][r], mx[r]);
        alpha[r] = __expf(m_run[mt][r] - mn);
        m_run[mt][r] = mn;
        rs[r] = 0.0f;
      }
      for (int nt = 0; nt < 4; ++nt)
        for (int r = 0; r < 8; ++r) {
          float p = __expf(sacc[mt][nt][r] - m_run[mt][r]);
          rs[r] += p;
          int irow = wave*32 + mt*16 + r + 8*g;     // per-wave private band
          sP[irow*72 + nt*16 + lr] = f2bf(p);
        }
      for (int off = 8; off >= 1; off >>= 1)
        for (int r = 0; r < 8; ++r)
          rs[r] += __shfl_xor(rs[r], off, 32);
      for (int r = 0; r < 8; ++r)
        l_run[mt][r] = l_run[mt][r] * alpha[r] + rs[r];
      for (int dt = 0; dt < 4; ++dt)
        for (int r = 0; r < 8; ++r)
          oacc[mt][dt][r] *= alpha[r];
    }

    // ---- O += P * V ----
    for (int mt = 0; mt < 2; ++mt) {
      for (int kk = 0; kk < 2; ++kk) {
        FragAB pf;
        int row = wave*32 + mt*16 + lr;
        pf.u[0] = *(const us8*)&sP[row*72 + kk*32 + g*8];
        pf.u[1] = *(const us8*)&sP[row*72 + kk*32 + 16 + g*8];
        for (int dt = 0; dt < 4; ++dt) {
          FragAB vf;
          int drow = dt*16 + lr;
          vf.u[0] = *(const us8*)&sVt[drow*72 + kk*32 + g*8];
          vf.u[1] = *(const us8*)&sVt[drow*72 + kk*32 + 16 + g*8];
          oacc[mt][dt] = __builtin_amdgcn_wmma_f32_16x16x32_bf16(
              false, pf.v, false, vf.v, (short)0, oacc[mt][dt], false, false);
        }
      }
    }
  }

  // ---- epilogue: O /= l, store bf16 ----
  for (int mt = 0; mt < 2; ++mt) {
    float rl[8];
    for (int r = 0; r < 8; ++r) rl[r] = 1.0f / l_run[mt][r];
    for (int dt = 0; dt < 4; ++dt)
      for (int r = 0; r < 8; ++r) {
        int i   = wave*32 + mt*16 + r + 8*g;
        int tok = q0 + i;
        O[(size_t)(b*SS + tok) * QW + h*HD + dt*16 + lr] =
            f2bf(oacc[mt][dt][r] * rl[r]);
      }
  }
}

// ---------------------------------------------------------------------------
extern "C" void kernel_launch(void* const* d_in, const int* in_sizes, int n_in,
                              void* d_out, int out_size, void* d_ws, size_t ws_size,
                              hipStream_t stream) {
  (void)in_sizes; (void)n_in; (void)out_size; (void)ws_size;
  const float* hs = (const float*)d_in[0];
  const float* Wq = (const float*)d_in[1];
  const float* Wk = (const float*)d_in[2];
  const float* Wv = (const float*)d_in[3];
  const float* Wo = (const float*)d_in[4];
  float* out = (float*)d_out;

  // bf16 workspace: Q(32MB) K(8MB) V(8MB) attn(32MB) = 80MB
  u16* Qb = (u16*)d_ws;
  u16* Kb = Qb + (size_t)MROWS * QW;
  u16* Vb = Kb + (size_t)MROWS * KVW;
  u16* Ab = Vb + (size_t)MROWS * KVW;

  dim3 blk(256);
  gemm_wmma<true,  false><<<dim3(QW/128,  MROWS/128), blk, 0, stream>>>(hs, Wq, Qb, MROWS, QW,  HH);
  gemm_wmma<true,  false><<<dim3(KVW/128, MROWS/128), blk, 0, stream>>>(hs, Wk, Kb, MROWS, KVW, HH);
  gemm_wmma<true,  false><<<dim3(KVW/128, MROWS/128), blk, 0, stream>>>(hs, Wv, Vb, MROWS, KVW, HH);

  int totq = MROWS * NH  * 32;
  int totk = MROWS * NKV * 32;
  rope_kernel<<<(totq + 255)/256, blk, 0, stream>>>(Qb, NH,  totq);
  rope_kernel<<<(totk + 255)/256, blk, 0, stream>>>(Kb, NKV, totk);

  attn_kernel<<<dim3(NH, NBLK, BB), blk, 0, stream>>>(Qb, Kb, Vb, Ab);

  gemm_wmma<false, true><<<dim3(QW/128, MROWS/128), blk, 0, stream>>>(Ab, Wo, out, MROWS, QW, HH);
}
